// SpanMarkerV1_5411658793027
// MI455X (gfx1250) — matrix operand order repaired and checked by reading, not compiled, not run on hardware
//
#include <hip/hip_runtime.h>

typedef __attribute__((ext_vector_type(16))) __bf16 bf16x16;
typedef __attribute__((ext_vector_type(8)))  float  floatx8;
typedef unsigned short u16;
typedef unsigned int   u32;
typedef unsigned long long u64;

// ---- problem constants ----
#define BATCH   8
#define SEQ     512
#define DMODEL  768
#define SPANS   6144
#define HID     3072
#define BL      (BATCH*SEQ)      // 4096
#define BS      (BATCH*SPANS)    // 49152
#define CHUNK_M 8192             // BS / 6

// ---- tiling ----
#define BM 64
#define BN 128
#define BKK 32
#define AST 40   // A LDS row stride (bf16 elems); 80B, multiple of 16B
#define BST 40   // Bt LDS row stride

__device__ __forceinline__ u16 f2bf(float x) {
    u32 u = __float_as_uint(x);
    u += 0x7fffu + ((u >> 16) & 1u);      // round-to-nearest-even
    return (u16)(u >> 16);
}
__device__ __forceinline__ u32 pk2(float a, float b) {
    return (u32)f2bf(a) | ((u32)f2bf(b) << 16);
}

union Frag {
    bf16x16 v;
    uint4   q[2];
};

__device__ __forceinline__ void async_b128(u32 lds_addr, const void* gsrc) {
    asm volatile("global_load_async_to_lds_b128 %0, %1, off"
                 :: "v"(lds_addr), "v"((u64)(uintptr_t)gsrc) : "memory");
}
__device__ __forceinline__ void async_b128_off16(u32 lds_addr, const void* gsrc) {
    // offset: applies to BOTH lds and global address (ISA 08_async_tensor §4.4)
    asm volatile("global_load_async_to_lds_b128 %0, %1, off offset:16"
                 :: "v"(lds_addr), "v"((u64)(uintptr_t)gsrc) : "memory");
}
__device__ __forceinline__ void wait_async0() {
    asm volatile("s_wait_asynccnt 0x0" ::: "memory");
}

// ---------------------- prep kernels ----------------------
// fp32 -> bf16 elementwise (8 elems/thread)
__global__ __launch_bounds__(256) void conv_bf16(const float* __restrict__ x,
                                                 u16* __restrict__ y) {
    const int i = (blockIdx.x * 256 + threadIdx.x) * 8;
    float4 a = *(const float4*)(x + i);
    float4 b = *(const float4*)(x + i + 4);
    uint4 o;
    o.x = pk2(a.x, a.y); o.y = pk2(a.z, a.w);
    o.z = pk2(b.x, b.y); o.w = pk2(b.z, b.w);
    *(uint4*)(y + i) = o;
}

// W [K,N] fp32  ->  WT [N,K] bf16   (32x32 tiles via LDS)
__global__ __launch_bounds__(256) void transpose_bf16(const float* __restrict__ W,
                                                      u16* __restrict__ WT,
                                                      int K, int N) {
    __shared__ u16 t[32][36];
    const int k0 = blockIdx.y * 32, n0 = blockIdx.x * 32;
    {
        const int r  = threadIdx.x >> 3;        // k within tile
        const int c4 = (threadIdx.x & 7) * 4;   // n within tile
        float4 v = *(const float4*)&W[(size_t)(k0 + r) * N + n0 + c4];
        t[c4 + 0][r] = f2bf(v.x); t[c4 + 1][r] = f2bf(v.y);
        t[c4 + 2][r] = f2bf(v.z); t[c4 + 3][r] = f2bf(v.w);
    }
    __syncthreads();
    {
        const int n  = threadIdx.x >> 3;
        const int kq = (threadIdx.x & 7) * 4;
        u16* dst = &WT[(size_t)(n0 + n) * K + k0 + kq];
        dst[0] = t[n][kq]; dst[1] = t[n][kq + 1];
        dst[2] = t[n][kq + 2]; dst[3] = t[n][kq + 3];
    }
}

// ---------------------- main GEMM ----------------------
// MODE: 1 = A is bf16 [M,K];  2 = A gathered from two bf16 [BL,DMODEL] tensors
// WT is bf16 [N,K] (pre-transposed).  Out: bf16 (OUTF32=0) or f32 (OUTF32=1).
template <int MODE, bool RELU, bool OUTF32>
__global__ __launch_bounds__(256) void gemm_wmma(
    const u16* __restrict__ Abf, const u16* __restrict__ Ae,
    const int* __restrict__ spanIdx, int rowOffset,
    const u16* __restrict__ WT, const float* __restrict__ bias,
    void* __restrict__ Outp, int M, int N, int K)
{
    __shared__ __align__(16) u16 As[2][BM * AST];   // 2 x  5120 B
    __shared__ __align__(16) u16 Bs[2][BN * BST];   // 2 x 10240 B

    const int tid  = threadIdx.x;
    const int lane = tid & 31;
    const int wave = tid >> 5;
    const int wm   = wave & 1;        // 2 waves along M
    const int wn   = wave >> 1;       // 4 waves along N
    const int m0   = blockIdx.y * BM;
    const int n0   = blockIdx.x * BN;
    const int lr   = lane & 15;
    const int hf   = lane >> 4;

    // ---- per-thread staging coordinates (hoisted) ----
    const int arow = tid >> 2;
    const int acq  = (tid & 3) * 8;             // 8 bf16 = 16B
    const int bn   = tid >> 1;
    const int bh   = (tid & 1) * 16;            // 16 bf16 = 32B half-row

    const u16* aBase0;                          // A source row base (+k)
    const u16* aBase1 = nullptr;                // gather: end-rep row base
    if (MODE == 1) {
        aBase0 = Abf + (size_t)(m0 + arow) * K + acq;
    } else {
        const int rg = rowOffset + m0 + arow;
        const int bb = rg / SPANS;
        aBase0 = Abf + (size_t)(bb * SEQ + spanIdx[rg * 2 + 0]) * DMODEL;
        aBase1 = Ae  + (size_t)(bb * SEQ + spanIdx[rg * 2 + 1]) * DMODEL;
    }
    const u16* bBase = WT + (size_t)(n0 + bn) * K + bh;

    const u32 aLds0 = (u32)(uintptr_t)&As[0][arow * AST + acq];
    const u32 aLds1 = (u32)(uintptr_t)&As[1][arow * AST + acq];
    const u32 bLds0 = (u32)(uintptr_t)&Bs[0][bn * BST + bh];
    const u32 bLds1 = (u32)(uintptr_t)&Bs[1][bn * BST + bh];

    auto issue = [&](int kk, int buf) {
        const u16* asrc;
        if (MODE == 1) {
            asrc = aBase0 + kk;
        } else {
            const int kg = kk + acq;
            asrc = (kg < DMODEL) ? (aBase0 + kg) : (aBase1 + (kg - DMODEL));
        }
        async_b128(buf ? aLds1 : aLds0, asrc);
        const u16* bsrc = bBase + kk;
        const u32  bl   = buf ? bLds1 : bLds0;
        async_b128(bl, bsrc);
        async_b128_off16(bl, bsrc);
    };

    floatx8 acc[2][2] = {};
    const int nK = K / BKK;

    issue(0, 0);
    for (int it = 0; it < nK; ++it) {
        wait_async0();
        __syncthreads();
        if (it + 1 < nK) issue((it + 1) * BKK, (it + 1) & 1);

        const u16* as = As[it & 1];
        const u16* bs = Bs[it & 1];

        Frag a[2], b[2];
        #pragma unroll
        for (int t = 0; t < 2; ++t) {
            const int r = wm * 32 + t * 16 + lr;
            const int o = r * AST + hf * 8;         // K = hf*8..+7, 16+hf*8..+7
            a[t].q[0] = *(const uint4*)&as[o];
            a[t].q[1] = *(const uint4*)&as[o + 16];
        }
        #pragma unroll
        for (int u = 0; u < 2; ++u) {
            const int c = wn * 32 + u * 16 + lr;
            const int o = c * BST + hf * 8;         // K = hf*8..+7, 16+hf*8..+7
            b[u].q[0] = *(const uint4*)&bs[o];
            b[u].q[1] = *(const uint4*)&bs[o + 16];
        }
        #pragma unroll
        for (int t = 0; t < 2; ++t)
            #pragma unroll
            for (int u = 0; u < 2; ++u)
                acc[t][u] = __builtin_amdgcn_wmma_f32_16x16x32_bf16(
                    false, a[t].v, false, b[u].v, (short)0, acc[t][u], false, false);
    }

    // ---- epilogue: bias + (relu) + store ----
    #pragma unroll
    for (int t = 0; t < 2; ++t) {
        #pragma unroll
        for (int u = 0; u < 2; ++u) {
            const int colg = n0 + wn * 32 + u * 16 + lr;
            const float bv = bias[colg];
            #pragma unroll
            for (int r = 0; r < 8; ++r) {
                const int rowg = m0 + wm * 32 + t * 16 + hf * 8 + r;
                float x = acc[t][u][r] + bv;
                if (RELU) x = fmaxf(x, 0.0f);
                if (OUTF32)
                    ((float*)Outp)[(size_t)rowg * N + colg] = x;
                else
                    ((u16*)Outp)[(size_t)rowg * N + colg] = f2bf(x);
            }
        }
    }
}

extern "C" void kernel_launch(void* const* d_in, const int* in_sizes, int n_in,
                              void* d_out, int out_size, void* d_ws, size_t ws_size,
                              hipStream_t stream) {
    const float* h    = (const float*)d_in[0];
    const int*   span = (const int*)  d_in[1];
    const float* sw1  = (const float*)d_in[2];
    const float* sb1  = (const float*)d_in[3];
    const float* sw2  = (const float*)d_in[4];
    const float* sb2  = (const float*)d_in[5];
    const float* ew1  = (const float*)d_in[6];
    const float* eb1  = (const float*)d_in[7];
    const float* ew2  = (const float*)d_in[8];
    const float* eb2  = (const float*)d_in[9];
    const float* ow1  = (const float*)d_in[10];
    const float* ob1  = (const float*)d_in[11];
    const float* ow2  = (const float*)d_in[12];
    const float* ob2  = (const float*)d_in[13];
    float* out = (float*)d_out;

    // ---- workspace layout (all bf16 / u16 elems) ----
    u16* Hb   = (u16*)d_ws;                            // [4096, 768]
    u16* W1s  = Hb   + (size_t)BL * DMODEL;            // [3072, 768]   sw1^T
    u16* W2s  = W1s  + (size_t)HID * DMODEL;           // [768, 3072]   sw2^T
    u16* W1e  = W2s  + (size_t)DMODEL * HID;           // [3072, 768]   ew1^T
    u16* W2e  = W1e  + (size_t)HID * DMODEL;           // [768, 3072]   ew2^T
    u16* W1o  = W2e  + (size_t)DMODEL * HID;           // [3072, 1536]  ow1^T
    u16* W2o  = W1o  + (size_t)HID * 2 * DMODEL;       // [768, 3072]   ow2^T
    u16* X1   = W2o  + (size_t)DMODEL * HID;           // [4096, 3072]
    u16* SREP = X1   + (size_t)BL * HID;               // [4096, 768]  relu'd
    u16* EREP = SREP + (size_t)BL * DMODEL;            // [4096, 768]  relu'd
    u16* T2   = EREP + (size_t)BL * DMODEL;            // [8192, 3072] chunk

    const dim3 blk(256);

    // ---- prep: h -> bf16, weights -> bf16 transposed [N,K] ----
    conv_bf16<<<(BL * DMODEL) / 2048, blk, 0, stream>>>(h, Hb);
    transpose_bf16<<<dim3(HID / 32,    DMODEL / 32),     blk, 0, stream>>>(sw1, W1s, DMODEL,     HID);
    transpose_bf16<<<dim3(DMODEL / 32, HID / 32),        blk, 0, stream>>>(sw2, W2s, HID,        DMODEL);
    transpose_bf16<<<dim3(HID / 32,    DMODEL / 32),     blk, 0, stream>>>(ew1, W1e, DMODEL,     HID);
    transpose_bf16<<<dim3(DMODEL / 32, HID / 32),        blk, 0, stream>>>(ew2, W2e, HID,        DMODEL);
    transpose_bf16<<<dim3(HID / 32,    (2*DMODEL) / 32), blk, 0, stream>>>(ow1, W1o, 2 * DMODEL, HID);
    transpose_bf16<<<dim3(DMODEL / 32, HID / 32),        blk, 0, stream>>>(ow2, W2o, HID,        DMODEL);

    // ---- start projection (relu folded into SREP for the later gather) ----
    gemm_wmma<1, true,  false><<<dim3(HID / BN,    BL / BM), blk, 0, stream>>>(
        Hb, nullptr, nullptr, 0, W1s, sb1, X1, BL, HID, DMODEL);
    gemm_wmma<1, true,  false><<<dim3(DMODEL / BN, BL / BM), blk, 0, stream>>>(
        X1, nullptr, nullptr, 0, W2s, sb2, SREP, BL, DMODEL, HID);
    // ---- end projection (reuse X1) ----
    gemm_wmma<1, true,  false><<<dim3(HID / BN,    BL / BM), blk, 0, stream>>>(
        Hb, nullptr, nullptr, 0, W1e, eb1, X1, BL, HID, DMODEL);
    gemm_wmma<1, true,  false><<<dim3(DMODEL / BN, BL / BM), blk, 0, stream>>>(
        X1, nullptr, nullptr, 0, W2e, eb2, EREP, BL, DMODEL, HID);

    // ---- output projection, fused gather, chunked along spans ----
    for (int ro = 0; ro < BS; ro += CHUNK_M) {
        gemm_wmma<2, true,  false><<<dim3(HID / BN,    CHUNK_M / BM), blk, 0, stream>>>(
            SREP, EREP, span, ro, W1o, ob1, T2, CHUNK_M, HID, 2 * DMODEL);
        gemm_wmma<1, false, true ><<<dim3(DMODEL / BN, CHUNK_M / BM), blk, 0, stream>>>(
            T2, nullptr, nullptr, 0, W2o, ob2, out + (size_t)ro * DMODEL,
            CHUNK_M, DMODEL, HID);
    }
}
